// Qwen3OmniMoeSparseMoeBlock_88424786690399
// MI455X (gfx1250) — matrix operand order repaired and checked by reading, not compiled, not run on hardware
//
#include <hip/hip_runtime.h>

#define T_TOK 8192
#define H_DIM 1024
#define I_DIM 512
#define E_NUM 32
#define K_TOP 4
#define TILE_M 64
#define NTHREADS 512

typedef unsigned short u16;
typedef __bf16 bf16_t;
typedef bf16_t v16bf __attribute__((ext_vector_type(16)));
typedef float v8f __attribute__((ext_vector_type(8)));

__device__ __forceinline__ u16 f2bf(float f) {
  unsigned u = __builtin_bit_cast(unsigned, f);
  u += 0x7FFFu + ((u >> 16) & 1u);   // round-to-nearest-even
  return (u16)(u >> 16);
}

union Frag32B { uint4 u[2]; v16bf v; };

__device__ __forceinline__ v16bf load_frag2(const u16* p0, const u16* p1) {
  Frag32B f;
  f.u[0] = *(const uint4*)p0;
  f.u[1] = *(const uint4*)p1;
  return f.v;
}

__device__ __forceinline__ float silu_fast(float s) {
  return s * __builtin_amdgcn_rcpf(1.f + __expf(-s));
}

// ---------------------------------------------------------------- kernel Z
__global__ void zero_counts_kernel(int* counts) {
  if (threadIdx.x < E_NUM) counts[threadIdx.x] = 0;
}

// ---------------------------------------------------------------- kernel W
// Pre-swizzle f32 weights [E][Kd][Nd] into bf16 B-fragment layout:
// out[(((e*NT + nt)*KK + kk)*32 + lane)*16 + hh] = W[e][kk*32 + (lane>>4)*16 + hh][nt*16 + (lane&15)]
__global__ __launch_bounds__(256) void swizzle_w_kernel(const float* __restrict__ W,
                                                        u16* __restrict__ out,
                                                        int Kd, int Nd) {
  int tid = blockIdx.x * 256 + threadIdx.x;
  const int KK = Kd >> 5, NT = Nd >> 4;
  const int lane = tid & 31;
  int rest = tid >> 5;
  const int kk = rest % KK; rest /= KK;
  const int nt = rest % NT;
  const int e  = rest / NT;
  if (e >= E_NUM) return;
  const int n  = nt * 16 + (lane & 15);
  const int kb = kk * 32 + ((lane >> 4) << 4);
  const float* src = W + ((size_t)e * Kd + kb) * Nd + n;
  union { u16 h[16]; uint4 u[2]; } pk;
#pragma unroll
  for (int j = 0; j < 16; ++j) pk.h[j] = f2bf(src[(size_t)j * Nd]);
  uint4* dst = (uint4*)(out + (size_t)tid * 16);
  dst[0] = pk.u[0];
  dst[1] = pk.u[1];
}

// ---------------------------------------------------------------- kernel A
// One wave32 per token: zero output row, emit bf16 x row, router logits
// (lane == expert), top-4 via shfl_xor argmax, scatter into expert buckets.
__global__ __launch_bounds__(256) void router_kernel(const float* __restrict__ x,
                                                     const float* __restrict__ rw,
                                                     float* __restrict__ dout,
                                                     u16* __restrict__ xb,
                                                     int* __restrict__ counts,
                                                     int* __restrict__ idxbuf) {
  const int wave = threadIdx.x >> 5;
  const int lane = threadIdx.x & 31;
  const int t = blockIdx.x * 8 + wave;
  const float* xr = x + (size_t)t * H_DIM;

  // zero the final-output row (we accumulate into it with atomics later)
  float* frow = dout + (size_t)t * H_DIM;
  const float4 z4 = make_float4(0.f, 0.f, 0.f, 0.f);
#pragma unroll
  for (int j = 0; j < 8; ++j) ((float4*)frow)[lane + j * 32] = z4;

  // bf16 copy of the x row
  u16* xbr = xb + (size_t)t * H_DIM;
#pragma unroll
  for (int c = 0; c < 8; ++c) {
    const int h = c * 128 + lane * 4;
    float4 v = *(const float4*)(xr + h);
    uint2 p;
    p.x = (unsigned)f2bf(v.x) | ((unsigned)f2bf(v.y) << 16);
    p.y = (unsigned)f2bf(v.z) | ((unsigned)f2bf(v.w) << 16);
    *(uint2*)(xbr + h) = p;
  }

  // router logit: lane <-> expert (E == 32 == wave32)
  float acc = 0.f;
#pragma unroll 4
  for (int h = 0; h < H_DIM; ++h)
    acc = __builtin_fmaf(xr[h], rw[h * E_NUM + lane], acc);
  dout[(size_t)T_TOK * H_DIM + (size_t)t * E_NUM + lane] = acc;

  // top-4 (softmax is monotonic; ties prefer lower index like lax.top_k)
  float v = acc;
  for (int it = 0; it < K_TOP; ++it) {
    float bv = v; int bl = lane;
#pragma unroll
    for (int off = 16; off > 0; off >>= 1) {
      float ov = __shfl_xor(bv, off, 32);
      int   ol = __shfl_xor(bl, off, 32);
      if (ov > bv || (ov == bv && ol < bl)) { bv = ov; bl = ol; }
    }
    if (lane == 0) {
      int pos = atomicAdd(counts + bl, 1);
      idxbuf[bl * T_TOK + pos] = t;
    }
    if (lane == bl) v = -__builtin_inff();
  }
}

// ---------------------------------------------------------------- kernel B
// 64-token tile, 512 threads (16 waves), K-phased X staging,
// N-tiles serialized so only 4 accumulators (32 VGPRs) cross each barrier.
#define XS_STRIDE 520    // halves per row: +4 dword bank shift per row
#define HS_STRIDE 520

__global__ __launch_bounds__(NTHREADS) void moe_expert_kernel(
    const u16* __restrict__ xb, const int* __restrict__ counts,
    const int* __restrict__ idxbuf,
    const u16* __restrict__ upB, const u16* __restrict__ dnB,
    const float* __restrict__ up_b, const float* __restrict__ down_b,
    float* __restrict__ out) {
  __shared__ u16 Xs[TILE_M * XS_STRIDE];    // 66,560 B (one K-half of X tile)
  __shared__ u16 Hs[TILE_M * HS_STRIDE];    // 66,560 B
  __shared__ int toks[TILE_M];

  const int e = blockIdx.y;
  const int Ne = counts[e];
  const int base = blockIdx.x * TILE_M;
  if (base >= Ne) return;           // uniform across block: EXEC stays full

  const int tid = threadIdx.x;
  if (tid < TILE_M)
    toks[tid] = (base + tid < Ne) ? idxbuf[e * T_TOK + base + tid] : -1;
  __syncthreads();

  const int w    = tid >> 5;        // 0..15
  const int lane = tid & 31;
  const int nlo  = lane & 15;
  const int hi   = lane >> 4;

  const u16* arow[4];
#pragma unroll
  for (int s = 0; s < 4; ++s) arow[s] = &Xs[(s * 16 + nlo) * XS_STRIDE + hi * 16];

  // ---------------- GEMM1: hmid[64 x 512] = silu(X @ Wup + bu)
  // q outer (serialize N-tiles), K-phase inner: only acc[4] live across barriers
#pragma unroll 1
  for (int q = 0; q < 2; ++q) {
    const int nt = w * 2 + q;
    v8f acc[4] = {};
#pragma unroll 1
    for (int ph = 0; ph < 2; ++ph) {
      // gather one K-half of the 64 token rows (bf16); zero-fill invalid rows
#pragma unroll
      for (int j = 0; j < 8; ++j) {
        const int li = tid + j * NTHREADS;   // 0..4095 uint4s
        const int m = li >> 6;               // row (64 uint4 per row)
        const int c = li & 63;
        const int tok = toks[m];
        uint4 v = {0u, 0u, 0u, 0u};
        if (tok >= 0)
          v = *(const uint4*)(xb + (size_t)tok * H_DIM + ph * 512 + c * 8);
        *(uint4*)&Xs[m * XS_STRIDE + c * 8] = v;
      }
      __syncthreads();

      const u16* bbase =
          upB + (size_t)((e * 32 + nt) * 32 + ph * 16) * 512 + lane * 16;
#pragma unroll 2
      for (int kk = 0; kk < 16; ++kk) {
        const int k0 = kk * 32;
        Frag32B B;
        const uint4* bp = (const uint4*)(bbase + (size_t)kk * 512);
        B.u[0] = bp[0]; B.u[1] = bp[1];
#pragma unroll
        for (int s = 0; s < 4; ++s) {
          v16bf A = load_frag2(arow[s] + k0, arow[s] + k0 + 8);
          acc[s] = __builtin_amdgcn_wmma_f32_16x16x32_bf16(
              false, A, false, B.v, (short)0, acc[s], false, false);
        }
      }
      __syncthreads();   // all reads of this K-half done before overwrite
    }
    // epilogue for this nt: bias + silu -> bf16 Hs
    const int n0 = nt * 16;
    const float bias = up_b[e * I_DIM + n0 + nlo];
#pragma unroll
    for (int s = 0; s < 4; ++s)
#pragma unroll
      for (int r = 0; r < 8; ++r)
        Hs[(s * 16 + r + hi * 8) * HS_STRIDE + n0 + nlo] =
            f2bf(silu_fast(acc[s][r] + bias));
  }
  __syncthreads();   // Hs complete before GEMM2 reads

  // ---------------- GEMM2: out[64 x 1024] = hmid @ Wdn + bd, atomic scatter
  const u16* brow[4];
#pragma unroll
  for (int s = 0; s < 4; ++s) brow[s] = &Hs[(s * 16 + nlo) * HS_STRIDE + hi * 16];

#pragma unroll 1
  for (int q = 0; q < 4; ++q) {
    const int nt = w * 4 + q;
    const int n0 = nt * 16;
    v8f acc[4] = {};
    const u16* bbase = dnB + (size_t)((e * 64 + nt) * 16) * 512 + lane * 16;
#pragma unroll 2
    for (int kk = 0; kk < 16; ++kk) {
      const int k0 = kk * 32;
      Frag32B B;
      const uint4* bp = (const uint4*)(bbase + (size_t)kk * 512);
      B.u[0] = bp[0]; B.u[1] = bp[1];
#pragma unroll
      for (int s = 0; s < 4; ++s) {
        v16bf A = load_frag2(brow[s] + k0, brow[s] + k0 + 8);
        acc[s] = __builtin_amdgcn_wmma_f32_16x16x32_bf16(
            false, A, false, B.v, (short)0, acc[s], false, false);
      }
    }
    const float bias = down_b[e * H_DIM + n0 + nlo];
#pragma unroll
    for (int s = 0; s < 4; ++s) {
#pragma unroll
      for (int r = 0; r < 8; ++r) {
        const int m = s * 16 + r + hi * 8;
        const int tok = toks[m];
        if (tok >= 0)
          __hip_atomic_fetch_add(out + (size_t)tok * H_DIM + n0 + nlo,
                                 acc[s][r] + bias, __ATOMIC_RELAXED,
                                 __HIP_MEMORY_SCOPE_AGENT);
      }
    }
  }
}

// ---------------------------------------------------------------- launch
extern "C" void kernel_launch(void* const* d_in, const int* in_sizes, int n_in,
                              void* d_out, int out_size, void* d_ws, size_t ws_size,
                              hipStream_t stream) {
  const float* x    = (const float*)d_in[0];
  const float* rw   = (const float*)d_in[1];
  const float* up_w = (const float*)d_in[2];
  const float* up_b = (const float*)d_in[3];
  const float* dn_w = (const float*)d_in[4];
  const float* dn_b = (const float*)d_in[5];
  float* out = (float*)d_out;

  char* ws = (char*)d_ws;
  // layout: counts(256B) | idx E*T*4 (1MB) | xb T*H*2 (16MB) | upB (32MB) | dnB (32MB)
  int* counts = (int*)ws;
  int* idxbuf = (int*)(ws + 256);
  u16* xb  = (u16*)(ws + 256 + (size_t)E_NUM * T_TOK * 4);
  u16* upB = (u16*)((char*)xb + (size_t)T_TOK * H_DIM * 2);
  u16* dnB = (u16*)((char*)upB + (size_t)E_NUM * H_DIM * I_DIM * 2);

  zero_counts_kernel<<<1, 32, 0, stream>>>(counts);
  // tasks = E*(Kd/32)*(Nd/16)*32 = 2^20 for both weight tensors
  swizzle_w_kernel<<<4096, 256, 0, stream>>>(up_w, upB, H_DIM, I_DIM);
  swizzle_w_kernel<<<4096, 256, 0, stream>>>(dn_w, dnB, I_DIM, H_DIM);
  router_kernel<<<T_TOK / 8, 256, 0, stream>>>(x, rw, out, xb, counts, idxbuf);
  moe_expert_kernel<<<dim3(T_TOK / TILE_M, E_NUM), NTHREADS, 0, stream>>>(
      xb, counts, idxbuf, upB, dnB, up_b, dn_b, out);
}
